// RelaxedRecursiveDoraLinear_18030272708749
// MI455X (gfx1250) — matrix operand order repaired and checked by reading, not compiled, not run on hardware
//
#include <hip/hip_runtime.h>

// ---------------------------------------------------------------------------
// RelaxedRecursiveDoraLinear for MI455X (gfx1250, wave32, WMMA).
//
// out = (scale-1)*(x@Wb^T + bias) + scale*((x@A^T)@B^T)
// scale[o] = mag[o] / || Wb[o,:] + 4 * (B@A)[o,:] ||
//
// Roofline: 275 GFLOP GEMM vs ~350MB (~15us) HBM -> matrix-compute bound.
// v_wmma_f32_16x16x32_bf16 (f32 -> bf16 RNE, f32 accumulate); LoRA folded
// into the same accumulator as one extra K=32 chunk (B pre-scaled s/(s-1)).
// Round 3: single-instruction v_cvt_pk_bf16_f32 packing; b128 LDS stores.
// ---------------------------------------------------------------------------

#define IN_F    4096
#define OUT_F   4096
#define RANK    16
#define M_TOT   8192            // BATCH*SEQ = 4*2048
#define SCALING 4.0f            // ALPHA / sqrt(RANK) = 16/4

typedef __attribute__((ext_vector_type(16))) __bf16 v16bf;
typedef __attribute__((ext_vector_type(2)))  __bf16 v2bf;
typedef __attribute__((ext_vector_type(8)))  float  v8f;

union Frag {
    uint4 u4[2];
    v16bf v;
};

// two f32 -> one dword of two bf16 (single v_cvt_pk_bf16_f32 when available)
__device__ __forceinline__ unsigned pack_bf2(float lo, float hi) {
#if __has_builtin(__builtin_amdgcn_cvt_pk_bf16_f32)
    v2bf p = __builtin_amdgcn_cvt_pk_bf16_f32(lo, hi);
#else
    v2bf p;
    p.x = (__bf16)lo;
    p.y = (__bf16)hi;
#endif
    return __builtin_bit_cast(unsigned, p);
}

// ---------------------------------------------------------------------------
// Kernel 1: scale[o] = mag/||Wb + 4*B@A||_row ; B'[o,r] = B[o,r]*s/(s-1)
// ---------------------------------------------------------------------------
__global__ __launch_bounds__(256)
void dora_scale_kernel(const float* __restrict__ w_base,
                       const float* __restrict__ lora_A,
                       const float* __restrict__ lora_B,
                       const float* __restrict__ magnitude,
                       const int*   __restrict__ loop_idx,
                       float* __restrict__ scale_out,
                       float* __restrict__ bms_out)
{
    const int li = loop_idx[0];
    const float* A   = lora_A    + (size_t)li * RANK  * IN_F;
    const float* Bm  = lora_B    + (size_t)li * OUT_F * RANK;
    const float* mag = magnitude + (size_t)li * OUT_F;

    __shared__ __attribute__((aligned(16))) float At[512][20]; // [k][r], pad for banks

    const int tid   = threadIdx.x;
    const int lane  = tid & 31;
    const int wid   = tid >> 5;
    const int oBase = blockIdx.x * 64;

    float acc[8];
#pragma unroll
    for (int j = 0; j < 8; ++j) acc[j] = 0.0f;

    for (int kc = 0; kc < IN_F; kc += 512) {
        __syncthreads();
        {
            const int r  = tid >> 4;
            const int k0 = tid & 15;
#pragma unroll 4
            for (int j = 0; j < 32; ++j) {
                const int kk = k0 + 16 * j;
                At[kk][r] = A[(size_t)r * IN_F + kc + kk];
            }
        }
        __syncthreads();
#pragma unroll 1
        for (int j = 0; j < 8; ++j) {
            const int o = oBase + wid * 8 + j;
            float bm[16];
#pragma unroll
            for (int r = 0; r < RANK; ++r) bm[r] = Bm[(size_t)o * RANK + r];
            const float* wrow = w_base + (size_t)o * IN_F + kc;
            float a = acc[j];
            for (int kk = lane; kk < 512; kk += 32) {
                float d = 0.0f;
                const float4* ap = (const float4*)&At[kk][0];
#pragma unroll
                for (int c = 0; c < 4; ++c) {
                    float4 v = ap[c];
                    d += bm[4*c+0]*v.x + bm[4*c+1]*v.y + bm[4*c+2]*v.z + bm[4*c+3]*v.w;
                }
                const float e = wrow[kk] + SCALING * d;
                a += e * e;
            }
            acc[j] = a;
        }
    }

#pragma unroll 1
    for (int j = 0; j < 8; ++j) {
        float a = acc[j];
#pragma unroll
        for (int sh = 16; sh >= 1; sh >>= 1) a += __shfl_xor(a, sh, 32);
        const int o = oBase + wid * 8 + j;
        const float s = mag[o] * __frsqrt_rn(a);
        float denom = s - 1.0f;
        if (fabsf(denom) < 1e-8f) denom = copysignf(1e-8f, denom);
        const float f = s / denom;
        if (lane == 0) scale_out[o] = s;
        if (lane < RANK) bms_out[(size_t)o * RANK + lane] = Bm[(size_t)o * RANK + lane] * f;
    }
}

// ---------------------------------------------------------------------------
// Kernel 2: t = x @ A^T   (M_TOT x RANK)
// ---------------------------------------------------------------------------
__global__ __launch_bounds__(256)
void lora_t_kernel(const float* __restrict__ x,
                   const float* __restrict__ lora_A,
                   const int*   __restrict__ loop_idx,
                   float* __restrict__ t_out)
{
    const int li = loop_idx[0];
    const float* A = lora_A + (size_t)li * RANK * IN_F;

    const int tid  = threadIdx.x;
    const int r    = tid & 15;
    const size_t m = (size_t)blockIdx.x * 16 + (tid >> 4);

    const float* xrow = x + m * IN_F;
    const float* arow = A + (size_t)r * IN_F;
    float acc = 0.0f;
    for (int k = 0; k < IN_F; k += 4) {
        const float4 xv = *(const float4*)(xrow + k);
        const float4 av = *(const float4*)(arow + k);
        acc += xv.x*av.x + xv.y*av.y + xv.z*av.z + xv.w*av.w;
    }
    t_out[m * RANK + r] = acc;
}

// ---------------------------------------------------------------------------
// Kernel 3: fused GEMM. 128x128 block tile, 8 waves of 32x64 (2x4 WMMA tiles).
// Software pipelined: preload chunk c+1 into VGPRs while WMMAs eat chunk c.
// Each thread stages 8 contiguous k-floats -> one ds_store_b128 per matrix.
// ---------------------------------------------------------------------------
struct StageRegs { float4 xv[4]; float4 wv[4]; };

__device__ __forceinline__ void preload_main(const float* __restrict__ x,
                                             const float* __restrict__ w,
                                             int mBlock, int nBlock, int kb,
                                             int tid, StageRegs& r)
{
#pragma unroll
    for (int i = 0; i < 2; ++i) {
        const int c    = tid + 256 * i;      // 0..511
        const int row  = c >> 2;             // 0..127
        const int seg8 = c & 3;              // 8-float segment in K=32
        const size_t xo = (size_t)(mBlock + row) * IN_F + kb + seg8 * 8;
        const size_t wo = (size_t)(nBlock + row) * IN_F + kb + seg8 * 8;
        r.xv[2*i]   = *(const float4*)(x + xo);
        r.xv[2*i+1] = *(const float4*)(x + xo + 4);
        r.wv[2*i]   = *(const float4*)(w + wo);
        r.wv[2*i+1] = *(const float4*)(w + wo + 4);
        if (kb + 32 < IN_F) {                // pull chunk c+2 toward caches
            __builtin_prefetch(x + xo + 32, 0, 1);
            __builtin_prefetch(w + wo + 32, 0, 1);
        }
    }
}

__device__ __forceinline__ void preload_lora(const float* __restrict__ tmat,
                                             const float* __restrict__ bms,
                                             int mBlock, int nBlock,
                                             int tid, StageRegs& r)
{
#pragma unroll
    for (int i = 0; i < 2; ++i) {
        const int c    = tid + 256 * i;
        const int row  = c >> 2;
        const int seg8 = c & 3;
        if (seg8 < 2) {                       // k 0..15 = t / B', k 16..31 = 0
            const float* tp = tmat + (size_t)(mBlock + row) * RANK + seg8 * 8;
            const float* bp = bms  + (size_t)(nBlock + row) * RANK + seg8 * 8;
            r.xv[2*i]   = *(const float4*)(tp);
            r.xv[2*i+1] = *(const float4*)(tp + 4);
            r.wv[2*i]   = *(const float4*)(bp);
            r.wv[2*i+1] = *(const float4*)(bp + 4);
        } else {
            r.xv[2*i] = r.xv[2*i+1] = make_float4(0.f, 0.f, 0.f, 0.f);
            r.wv[2*i] = r.wv[2*i+1] = make_float4(0.f, 0.f, 0.f, 0.f);
        }
    }
}

__device__ __forceinline__ void store_stage(unsigned char* xsm, unsigned char* wsm,
                                            int tid, const StageRegs& r)
{
#pragma unroll
    for (int i = 0; i < 2; ++i) {
        const int c    = tid + 256 * i;
        const int row  = c >> 2;
        const int seg8 = c & 3;
        const uint4 px = make_uint4(pack_bf2(r.xv[2*i].x,   r.xv[2*i].y),
                                    pack_bf2(r.xv[2*i].z,   r.xv[2*i].w),
                                    pack_bf2(r.xv[2*i+1].x, r.xv[2*i+1].y),
                                    pack_bf2(r.xv[2*i+1].z, r.xv[2*i+1].w));
        const uint4 pw = make_uint4(pack_bf2(r.wv[2*i].x,   r.wv[2*i].y),
                                    pack_bf2(r.wv[2*i].z,   r.wv[2*i].w),
                                    pack_bf2(r.wv[2*i+1].x, r.wv[2*i+1].y),
                                    pack_bf2(r.wv[2*i+1].z, r.wv[2*i+1].w));
        *(uint4*)(xsm + row * 64 + seg8 * 16) = px;
        *(uint4*)(wsm + row * 64 + seg8 * 16) = pw;
    }
}

__device__ __forceinline__ void compute_wmma(const unsigned char* xsm,
                                             const unsigned char* wsm,
                                             int m0, int n0, int lane,
                                             v8f acc[2][4])
{
    Frag af[2], bfr[4];
    const int aByte = (lane >= 16) ? 16 : 0;   // K 8..15 half for hi lanes
    const int bByte = (lane >= 16) ? 16 : 0;   // K 8..15 half for hi lanes
#pragma unroll
    for (int mt = 0; mt < 2; ++mt) {
        const unsigned char* p = xsm + (m0 + mt * 16 + (lane & 15)) * 64 + aByte;
        af[mt].u4[0] = *(const uint4*)(p);        // K 0..7   (or 8..15)
        af[mt].u4[1] = *(const uint4*)(p + 32);   // K 16..23 (or 24..31)
    }
#pragma unroll
    for (int nt = 0; nt < 4; ++nt) {
        const unsigned char* p = wsm + (n0 + nt * 16 + (lane & 15)) * 64 + bByte;
        bfr[nt].u4[0] = *(const uint4*)(p);       // K 0..7   (or 8..15)
        bfr[nt].u4[1] = *(const uint4*)(p + 32);  // K 16..23 (or 24..31)
    }
#pragma unroll
    for (int mt = 0; mt < 2; ++mt)
#pragma unroll
        for (int nt = 0; nt < 4; ++nt)
            acc[mt][nt] = __builtin_amdgcn_wmma_f32_16x16x32_bf16(
                false, af[mt].v, false, bfr[nt].v,
                (short)0, acc[mt][nt], false, false);
}

__global__ __launch_bounds__(256)
void dora_gemm_kernel(const float* __restrict__ x,
                      const float* __restrict__ w_base,
                      const float* __restrict__ bias,
                      const float* __restrict__ scale,
                      const float* __restrict__ bms,    // [OUT_F][RANK] pre-scaled
                      const float* __restrict__ tmat,   // [M_TOT][RANK]
                      float* __restrict__ out)
{
    __shared__ __attribute__((aligned(16))) unsigned char smem[16384];
    unsigned char* xsm = smem;          // 128 x 32 bf16, row stride 64B
    unsigned char* wsm = smem + 8192;   // 128 x 32 bf16, row stride 64B

    const int tid    = threadIdx.x;
    const int lane   = tid & 31;
    const int wid    = tid >> 5;
    const int mBlock = blockIdx.y * 128;
    const int nBlock = blockIdx.x * 128;
    const int m0     = (wid & 3) * 32;   // 4 waves along M
    const int n0     = (wid >> 2) * 64;  // 2 waves along N

    v8f acc[2][4] = {};
    StageRegs sr;

    preload_main(x, w_base, mBlock, nBlock, 0, tid, sr);

    // branch-free hot loop: chunks 0..126 computed, chunk c+1 preloaded
#pragma unroll 1
    for (int c = 0; c < 127; ++c) {
        __syncthreads();
        store_stage(xsm, wsm, tid, sr);
        __syncthreads();
        preload_main(x, w_base, mBlock, nBlock, (c + 1) * 32, tid, sr);
        compute_wmma(xsm, wsm, m0, n0, lane, acc);
    }

    // chunk 127 + LoRA chunk, fully peeled
    __syncthreads();
    store_stage(xsm, wsm, tid, sr);
    __syncthreads();
    preload_lora(tmat, bms, mBlock, nBlock, tid, sr);
    compute_wmma(xsm, wsm, m0, n0, lane, acc);

    __syncthreads();
    store_stage(xsm, wsm, tid, sr);
    __syncthreads();
    compute_wmma(xsm, wsm, m0, n0, lane, acc);

    // ---- epilogue: out = (s-1) * (acc + bias) ----
    const int col = lane & 15;
    const int hi  = lane >> 4;
#pragma unroll
    for (int nt = 0; nt < 4; ++nt) {
        const int o   = nBlock + n0 + nt * 16 + col;
        const float s = scale[o];
        const float b = bias[o];
        const float sm1 = s - 1.0f;
#pragma unroll
        for (int mt = 0; mt < 2; ++mt) {
            const size_t gm = (size_t)(mBlock + m0 + mt * 16 + hi * 8);
#pragma unroll
            for (int r = 0; r < 8; ++r)
                out[(gm + r) * OUT_F + o] = sm1 * (acc[mt][nt][r] + b);
        }
    }
}

// ---------------------------------------------------------------------------
extern "C" void kernel_launch(void* const* d_in, const int* in_sizes, int n_in,
                              void* d_out, int out_size, void* d_ws, size_t ws_size,
                              hipStream_t stream) {
    const float* x         = (const float*)d_in[0];
    const float* w_base    = (const float*)d_in[1];
    const float* bias      = (const float*)d_in[2];
    const float* lora_A    = (const float*)d_in[3];
    const float* lora_B    = (const float*)d_in[4];
    const float* magnitude = (const float*)d_in[5];
    const int*   loop_idx  = (const int*)d_in[6];
    float* out = (float*)d_out;

    // workspace: scale[4096] | B'[4096*16] | t[8192*16]  (~0.8 MB)
    float* scale = (float*)d_ws;
    float* bms   = scale + OUT_F;
    float* tmat  = bms + (size_t)OUT_F * RANK;

    hipLaunchKernelGGL(dora_scale_kernel, dim3(OUT_F / 64), dim3(256), 0, stream,
                       w_base, lora_A, lora_B, magnitude, loop_idx, scale, bms);
    hipLaunchKernelGGL(lora_t_kernel, dim3(M_TOT / 16), dim3(256), 0, stream,
                       x, lora_A, loop_idx, tmat);
    hipLaunchKernelGGL(dora_gemm_kernel, dim3(OUT_F / 128, M_TOT / 128), dim3(256),
                       0, stream, x, w_base, bias, scale, bms, tmat, out);
}
